// mLSTMCell_39006892982942
// MI455X (gfx1250) — matrix operand order repaired and checked by reading, not compiled, not run on hardware
//
#include <hip/hip_runtime.h>
#include <hip/hip_bf16.h>
#include <math.h>

// ---------------------------------------------------------------------------
// mLSTM: hoist all input projections into one f16 WMMA GEMM
//   X[8192x1024] * W^T[1024x4352]  (4352 = 16 heads * 272; 258 real rows/head)
// then run the sequential 64x64-matrix-memory scan with c held in VGPRs.
// ---------------------------------------------------------------------------

#define H   16
#define HS  64
#define D   1024
#define Bb  8
#define T   1024
#define M_  (Bb * T)          // 8192
#define Rpad 272              // per-head padded row count (multiple of 16)
#define N_  (H * Rpad)        // 4352

typedef __attribute__((ext_vector_type(16))) _Float16     v16h;
typedef __attribute__((ext_vector_type(8)))  float        v8f;
typedef __attribute__((ext_vector_type(4)))  unsigned int v4u;

union AFrag { v16h h; v4u u[2]; };
union BFrag { v16h h; v4u u[2]; };

// ---------------- conversion kernels -------------------------------------

__global__ __launch_bounds__(256) void convert_x(const float* __restrict__ xs,
                                                 _Float16* __restrict__ x16) {
    size_t base = ((size_t)blockIdx.x * 256 + threadIdx.x) * 4;
    if (base >= (size_t)M_ * D) return;
    float4 v = *(const float4*)(xs + base);
    x16[base + 0] = (_Float16)v.x;
    x16[base + 1] = (_Float16)v.y;
    x16[base + 2] = (_Float16)v.z;
    x16[base + 3] = (_Float16)v.w;
}

__global__ __launch_bounds__(256) void convert_w(const float* __restrict__ if_w,
                                                 const float* __restrict__ o_w,
                                                 const float* __restrict__ kvq_w,
                                                 _Float16* __restrict__ w16) {
    size_t base = ((size_t)blockIdx.x * 256 + threadIdx.x) * 4;
    if (base >= (size_t)N_ * D) return;
    int n = (int)(base / D);
    int k = (int)(base % D);
    int h = n / Rpad;
    int r = n % Rpad;
    const float* src = nullptr;
    if (r < 2)        src = if_w  + ((size_t)(h * 2   + r)        * D + k);
    else if (r < 66)  src = o_w   + ((size_t)(h * 64  + (r - 2))  * D + k);
    else if (r < 258) src = kvq_w + ((size_t)(h * 192 + (r - 66)) * D + k);
    if (src) {
        float4 v = *(const float4*)src;
        w16[base + 0] = (_Float16)v.x;
        w16[base + 1] = (_Float16)v.y;
        w16[base + 2] = (_Float16)v.z;
        w16[base + 3] = (_Float16)v.w;
    } else {
        w16[base + 0] = (_Float16)0.f;
        w16[base + 1] = (_Float16)0.f;
        w16[base + 2] = (_Float16)0.f;
        w16[base + 3] = (_Float16)0.f;
    }
}

// ---------------- WMMA GEMM: G[m][n] = sum_k X[m][k] * W[n][k] -------------
// block tile 128(M) x 128(N), 8 waves in a 4(M) x 2(N) grid,
// each wave: 2x4 tiles of v_wmma_f32_16x16x32_f16, K staged 32 at a time.

__global__ __launch_bounds__(256) void gemm_wmma(const _Float16* __restrict__ A,
                                                 const _Float16* __restrict__ Bw,
                                                 float* __restrict__ G) {
    __shared__ __align__(16) _Float16 sA[128 * 32];
    __shared__ __align__(16) _Float16 sB[128 * 32];

    const int n0    = blockIdx.x * 128;
    const int m0    = blockIdx.y * 128;
    const int tid   = threadIdx.x;
    const int lane  = tid & 31;
    const int wave  = tid >> 5;
    const int waveM = wave & 3;     // 0..3
    const int waveN = wave >> 2;    // 0..1
    const int lr    = lane & 15;
    const int g     = lane >> 4;    // half-wave id

    v8f acc[2][4] = {};

    for (int kc = 0; kc < D; kc += 32) {
        // cooperative tile load: 128 rows x 32 halfs each for A and B
        #pragma unroll
        for (int s = 0; s < 2; ++s) {
            int seg = tid + s * 256;       // 0..511
            int row = seg >> 2;
            int c16 = seg & 3;             // 8-half (16B) segment
            *(v4u*)(sA + row * 32 + c16 * 8) =
                *(const v4u*)(A + (size_t)(m0 + row) * D + kc + c16 * 8);
            *(v4u*)(sB + row * 32 + c16 * 8) =
                *(const v4u*)(Bw + (size_t)(n0 + row) * D + kc + c16 * 8);
        }
        __syncthreads();

        // A fragments: lane holds row M=lr; elems 0-7 = K 8g..8g+7,
        //              elems 8-15 = K 16+8g..16+8g+7   (ISA 7.12.2)
        AFrag af[2];
        #pragma unroll
        for (int mi = 0; mi < 2; ++mi) {
            int row = waveM * 32 + mi * 16 + lr;
            af[mi].u[0] = *(const v4u*)(sA + row * 32 + g * 8);
            af[mi].u[1] = *(const v4u*)(sA + row * 32 + 16 + g * 8);
        }
        // B fragments: lane holds column N=lr; elems 0-7 = K 8g..8g+7, elems 8-15 = K 16+8g..16+8g+7
        BFrag bf[4];
        #pragma unroll
        for (int ni = 0; ni < 4; ++ni) {
            int n = waveN * 64 + ni * 16 + lr;
            bf[ni].u[0] = *(const v4u*)(sB + n * 32 + g * 8);
            bf[ni].u[1] = *(const v4u*)(sB + n * 32 + 16 + g * 8);
        }

        #pragma unroll
        for (int mi = 0; mi < 2; ++mi)
            #pragma unroll
            for (int ni = 0; ni < 4; ++ni)
                acc[mi][ni] = __builtin_amdgcn_wmma_f32_16x16x32_f16(
                    false, af[mi].h, false, bf[ni].h,
                    (short)0, acc[mi][ni], false, false);
        __syncthreads();
    }

    // store: C/D layout — VGPR r, lanes 0-15 hold M=r, lanes 16-31 hold M=8+r
    #pragma unroll
    for (int mi = 0; mi < 2; ++mi)
        #pragma unroll
        for (int ni = 0; ni < 4; ++ni)
            #pragma unroll
            for (int r = 0; r < 8; ++r) {
                int m = m0 + waveM * 32 + mi * 16 + g * 8 + r;
                int n = n0 + waveN * 64 + ni * 16 + lr;
                G[(size_t)m * N_ + n] = acc[mi][ni][r];
            }
}

// ---------------- sequential mLSTM scan -----------------------------------
// one 64-thread block per (batch, head); thread i owns row i of c (in VGPRs),
// n[i], and output element i. k/q broadcast via LDS, denom via LDS reduction,
// next-step gates software-prefetched.

__global__ __launch_bounds__(64) void mlstm_scan(const float* __restrict__ G,
                                                 const float* __restrict__ if_b,
                                                 const float* __restrict__ o_b,
                                                 const float* __restrict__ kvq_b,
                                                 float* __restrict__ out) {
    const int bh  = blockIdx.x;        // 0..127
    const int b   = bh >> 4;
    const int h   = bh & 15;
    const int tid = threadIdx.x;       // 0..63

    __shared__ __align__(16) float lds_k[64];
    __shared__ __align__(16) float lds_q[64];
    __shared__ __align__(16) float lds_red[64];

    float c[64];
    #pragma unroll
    for (int j = 0; j < 64; ++j) c[j] = 0.f;
    float n_i = 0.f;

    const float ib = if_b[h * 2 + 0];
    const float fb = if_b[h * 2 + 1];
    const float ob = o_b[h * 64 + tid];
    const float kb = kvq_b[(h * 3 + 0) * 64 + tid];
    const float vb = kvq_b[(h * 3 + 1) * 64 + tid];
    const float qb = kvq_b[(h * 3 + 2) * 64 + tid];

    const float* g0   = G + (size_t)(b * T) * N_ + h * Rpad;
    float*       outp = out + (size_t)(b * T) * (H * HS) + h * HS + tid;

    // prefetch gates for t = 0
    float p_if0 = g0[0],        p_if1 = g0[1];
    float p_o   = g0[2 + tid],  p_k   = g0[66 + tid];
    float p_v   = g0[130 + tid], p_q  = g0[194 + tid];

    for (int t = 0; t < T; ++t) {
        float if0 = p_if0, if1 = p_if1;
        float orw = p_o, krw = p_k, vrw = p_v, qrw = p_q;
        if (t + 1 < T) {   // issue next-step loads; latency hides under compute
            const float* gp = g0 + (size_t)(t + 1) * N_;
            p_if0 = gp[0];        p_if1 = gp[1];
            p_o   = gp[2 + tid];  p_k   = gp[66 + tid];
            p_v   = gp[130 + tid]; p_q  = gp[194 + tid];
        }

        float ig = __expf(if0 + ib);
        float fg = __expf(if1 + fb);
        float kk = krw * 0.125f + kb;          // K_SCALE = 1/sqrt(64)
        float vv = vrw + vb;
        float qq = qrw + qb;
        float oo = 1.f / (1.f + __expf(-(orw + ob)));

        lds_k[tid] = kk;
        lds_q[tid] = qq;
        n_i = fg * n_i + ig * kk;
        lds_red[tid] = n_i * qq;
        __syncthreads();

        float a  = ig * vv;
        float hn = 0.f;
        #pragma unroll
        for (int j = 0; j < 64; j += 4) {
            float4 kv = *(const float4*)(lds_k + j);
            float4 qv = *(const float4*)(lds_q + j);
            c[j + 0] = fg * c[j + 0] + a * kv.x; hn += c[j + 0] * qv.x;
            c[j + 1] = fg * c[j + 1] + a * kv.y; hn += c[j + 1] * qv.y;
            c[j + 2] = fg * c[j + 2] + a * kv.z; hn += c[j + 2] * qv.z;
            c[j + 3] = fg * c[j + 3] + a * kv.w; hn += c[j + 3] * qv.w;
        }
        float den = 0.f;
        #pragma unroll
        for (int j = 0; j < 64; j += 4) {
            float4 rv = *(const float4*)(lds_red + j);
            den += rv.x + rv.y + rv.z + rv.w;
        }
        den = fabsf(den);
        den = den < 1.f ? 1.f : den;

        outp[(size_t)t * (H * HS)] = oo * hn / den;
        __syncthreads();   // protect LDS before next step's writes
    }
}

// ---------------- launcher -------------------------------------------------

extern "C" void kernel_launch(void* const* d_in, const int* in_sizes, int n_in,
                              void* d_out, int out_size, void* d_ws, size_t ws_size,
                              hipStream_t stream) {
    const float* xs    = (const float*)d_in[0];  // [B,T,D]
    const float* if_w  = (const float*)d_in[1];  // [H,2,D]
    const float* o_w   = (const float*)d_in[2];  // [H,HS,D]
    const float* kvq_w = (const float*)d_in[3];  // [H,3*HS,D]
    const float* if_b  = (const float*)d_in[4];  // [H,2]
    const float* o_b   = (const float*)d_in[5];  // [H,HS]
    const float* kvq_b = (const float*)d_in[6];  // [H,3,HS]
    float*       out   = (float*)d_out;          // [B,T,H*HS]

    char* ws = (char*)d_ws;
    const size_t x16_bytes = (size_t)M_ * D * sizeof(_Float16);   // 16.8 MB
    const size_t w16_bytes = (size_t)N_ * D * sizeof(_Float16);   //  8.9 MB
    _Float16* x16 = (_Float16*)ws;
    _Float16* w16 = (_Float16*)(ws + x16_bytes);
    float*    G   = (float*)(ws + x16_bytes + w16_bytes);         // 142.6 MB

    // 1) convert inputs to f16
    convert_x<<<(M_ * (size_t)D / 4 + 255) / 256, 256, 0, stream>>>(xs, x16);
    convert_w<<<((size_t)N_ * D / 4 + 255) / 256, 256, 0, stream>>>(if_w, o_w, kvq_w, w16);

    // 2) all projections as one WMMA GEMM
    dim3 gGrid(N_ / 128, M_ / 128);   // (34, 64)
    gemm_wmma<<<gGrid, 256, 0, stream>>>(x16, w16, G);

    // 3) sequential recurrence, one block per (batch, head)
    mlstm_scan<<<Bb * H, 64, 0, stream>>>(G, if_b, o_b, kvq_b, out);
}
